// ReasoningMultiHeadAttention_58394375356558
// MI455X (gfx1250) — matrix operand order repaired and checked by reading, not compiled, not run on hardware
//
#include <hip/hip_runtime.h>
#include <hip/hip_bf16.h>

#define H 1024
#define HEADS 16
#define DH 64
#define BATCH 2
#define SEQ 2048
#define BS (BATCH * SEQ)

typedef __attribute__((ext_vector_type(16))) __bf16 v16bf;
typedef __attribute__((ext_vector_type(8)))  float  v8f;
typedef unsigned int v4u  __attribute__((ext_vector_type(4)));
typedef int          v8i_t __attribute__((ext_vector_type(8)));
typedef int          v4i_t __attribute__((ext_vector_type(4)));

union Frag {
    unsigned int u[8];
    v16bf v;
};

__device__ __forceinline__ unsigned short f2bf(float x) {
    unsigned int u = __builtin_bit_cast(unsigned int, x);
    unsigned int r = u + 0x7FFFu + ((u >> 16) & 1u);
    return (unsigned short)(r >> 16);
}

__device__ __forceinline__ v8f bfwmma(const Frag& a, const Frag& b, v8f c) {
    return __builtin_amdgcn_wmma_f32_16x16x32_bf16(
        false, a.v, false, b.v, (short)0, c, false, false);
}

// A-fragment (16x32 bf16): lane holds row (lane&15); K pattern per ISA table.
// `base` must point at rowStart + k0 + 8*(lane>=16).
__device__ __forceinline__ void load_a(Frag& f, const unsigned short* __restrict__ base) {
#pragma unroll
    for (int v = 0; v < 8; ++v) {
        const int kb = (v < 4) ? (2 * v) : (8 + 2 * v);
        f.u[v] = *(const unsigned int*)(base + kb);
    }
}

__device__ __forceinline__ void load_b(Frag& f, const unsigned short* base, int ld) {
#pragma unroll
    for (int v = 0; v < 8; ++v) {
        const int kb = (v < 4) ? (2 * v) : (8 + 2 * v);
        f.u[v] = (unsigned int)base[kb * ld] | ((unsigned int)base[(kb + 1) * ld] << 16);
    }
}

// ---- Tensor Data Mover: async 2D tile (bf16) global -> LDS ----
// tile_w elems contiguous per row, tile_h rows, row stride in elements.
__device__ __forceinline__ void tdm_load_tile_2d(unsigned lds_off,
                                                 const unsigned short* gptr,
                                                 unsigned tile_w, unsigned tile_h,
                                                 unsigned row_stride)
{
    unsigned long long ga = (unsigned long long)(uintptr_t)gptr;
    v4u g0;
    g0[0] = 1u;                                                  // count=1 (valid user D#)
    g0[1] = lds_off;                                             // LDS byte address
    g0[2] = (unsigned)(ga & 0xFFFFFFFFull);                      // global_addr[31:0]
    g0[3] = (unsigned)((ga >> 32) & 0x01FFFFFFull) | (2u << 30); // addr[56:32] | type=2
    const unsigned td0 = 0x40000000u, td1 = 0x40000000u;         // huge dims: no OOB clip
    v8i_t g1;
    g1[0] = (int)(1u << 16);                                     // data_size=2B, mask=0
    g1[1] = (int)((td0 & 0xFFFFu) << 16);                        // tensor_dim0 lo16
    g1[2] = (int)((td0 >> 16) | ((td1 & 0xFFFFu) << 16));        // dim0 hi / dim1 lo
    g1[3] = (int)((td1 >> 16) | (tile_w << 16));                 // dim1 hi | tile_dim0
    g1[4] = (int)(tile_h & 0xFFFFu);                             // tile_dim1 (dim2=0)
    g1[5] = (int)row_stride;                                     // tensor_dim0_stride lo32
    g1[6] = 0;
    g1[7] = 0;
    v4i_t z4 = {0, 0, 0, 0};
#if __clang_major__ >= 23
    v8i_t z8 = {0, 0, 0, 0, 0, 0, 0, 0};
    __builtin_amdgcn_tensor_load_to_lds(g0, g1, z4, z4, z8, 0);
#else
    __builtin_amdgcn_tensor_load_to_lds(g0, g1, z4, z4, 0);
#endif
}

// ---------------- conversion kernels ----------------

__global__ void k_cvt_x(const float* __restrict__ x, unsigned short* __restrict__ xh, int n) {
    int i = blockIdx.x * blockDim.x + threadIdx.x;
    if (i < n) xh[i] = f2bf(x[i]);
}

// Wt[k*H + n] = bf16(W[n*H + k])
__global__ void k_cvt_w(const float* __restrict__ W, unsigned short* __restrict__ Wt) {
    int n = blockIdx.x;
    for (int k = threadIdx.x; k < H; k += blockDim.x)
        Wt[k * H + n] = f2bf(W[n * H + k]);
}

// ---------------- fused QKV(+reasoning) GEMM with TDM staging ----------------

__global__ __launch_bounds__(256) void k_qkv(
    const unsigned short* __restrict__ xh,
    const unsigned short* __restrict__ wq, const unsigned short* __restrict__ wk,
    const unsigned short* __restrict__ wvw, const unsigned short* __restrict__ wqr,
    const unsigned short* __restrict__ wkr,
    const float* __restrict__ bq, const float* __restrict__ bk,
    const float* __restrict__ bv, const float* __restrict__ bqr,
    const float* __restrict__ bkr,
    const float* __restrict__ rmask,
    unsigned short* __restrict__ qb, unsigned short* __restrict__ kt,
    unsigned short* __restrict__ vb)
{
    __shared__ unsigned short wtile[2][5][32 * 16];   // 10 KB, double buffered
    const int lane = threadIdx.x & 31;
    const int wvid = threadIdx.x >> 5;
    const int mblk = blockIdx.x >> 6;                 // 32 blocks of M=128
    const int nt = blockIdx.x & 63;                   // 64 n-tiles of 16
    const int m0 = (mblk * 8 + wvid) * 16;
    const int n0 = nt * 16;
    const int lrow = lane & 15;
    const int halfSel = (lane >> 4) & 1;
    const int arow = m0 + lrow;

    const unsigned short* wsrc[5] = {wq, wk, wvw, wqr, wkr};
    v8f acc[5] = {v8f{}, v8f{}, v8f{}, v8f{}, v8f{}};

    auto issue = [&](int par, int k0) {
#pragma unroll
        for (int j = 0; j < 5; ++j)
            tdm_load_tile_2d((unsigned)(uintptr_t)&wtile[par][j][0],
                             wsrc[j] + (size_t)k0 * H + n0, 16, 32, H);
    };

    if (threadIdx.x < 32) issue(0, 0);

    for (int it = 0; it < 32; ++it) {
        const int k0 = it * 32;
        if (threadIdx.x < 32) {
            if (it + 1 < 32) {
                issue((it + 1) & 1, k0 + 32);
                __builtin_amdgcn_s_wait_tensorcnt((short)5);   // retire current buffer
            } else {
                __builtin_amdgcn_s_wait_tensorcnt((short)0);
            }
        }
        __syncthreads();

        Frag a;
        load_a(a, xh + (size_t)arow * H + k0 + halfSel * 8);
        const int par = it & 1;
#pragma unroll
        for (int j = 0; j < 5; ++j) {
            Frag b;
            load_b(b, &wtile[par][j][halfSel * 8 * 16 + lrow], 16);
            acc[j] = bfwmma(a, b, acc[j]);
        }
        __syncthreads();
    }

    const int col = n0 + lrow;
    const float vbq = bq[col], vbk = bk[col], vbv = bv[col];
    const float vbqr = bqr[col], vbkr = bkr[col];
    const int hh = col >> 6, dd = col & 63;

#pragma unroll
    for (int i = 0; i < 8; ++i) {
        const int r = m0 + i + halfSel * 8;           // flat token index
        const float rm = rmask[r];
        const int bi = r >> 11, s = r & (SEQ - 1);
        const float qv = (acc[0][i] + vbq) * (1.f - rm) + (acc[3][i] + vbqr) * rm;
        const float kv = (acc[1][i] + vbk) * (1.f - rm) + (acc[4][i] + vbkr) * rm;
        const float vv = acc[2][i] + vbv;
        const size_t bh = (size_t)(bi * HEADS + hh);
        qb[(bh * SEQ + s) * DH + dd] = f2bf(qv);
        kt[(bh * DH + dd) * SEQ + s] = f2bf(kv);      // K stored transposed [b,h,d,s]
        vb[(bh * SEQ + s) * DH + dd] = f2bf(vv);
    }
}

// ---------------- flash attention with TDM-staged K^T / V tiles ----------------

__global__ __launch_bounds__(256) void k_attn(
    const unsigned short* __restrict__ qb, const unsigned short* __restrict__ kt,
    const unsigned short* __restrict__ vbg, const int* __restrict__ amask,
    unsigned short* __restrict__ ao)
{
    __shared__ unsigned short kt_s[2][64 * 32];       // 8 KB
    __shared__ unsigned short v_s[2][32 * 64];        // 8 KB
    __shared__ unsigned short pbuf[8][16 * 32];       // 8 KB (per-wave P swizzle)

    const int lane = threadIdx.x & 31;
    const int wvid = threadIdx.x >> 5;
    const int gid = blockIdx.x;                       // 512 blocks
    const int bh = gid >> 4;                          // b*HEADS + h (shared by block)
    const int qt = (gid & 15) * 8 + wvid;             // 0..127
    const int qi0 = qt * 16;
    const int bi = bh >> 4;
    const int hh = bh & 15;
    const int lrow = lane & 15;
    const int halfSel = (lane >> 4) & 1;

    const unsigned short* qbase = qb  + (size_t)bh * SEQ * DH;
    const unsigned short* ktb   = kt  + (size_t)bh * DH * SEQ;
    const unsigned short* vbase = vbg + (size_t)bh * SEQ * DH;

    Frag aq0, aq1;
    load_a(aq0, qbase + (size_t)(qi0 + lrow) * DH + 0  + halfSel * 8);
    load_a(aq1, qbase + (size_t)(qi0 + lrow) * DH + 32 + halfSel * 8);

    float mrun[8], lrun[8];
#pragma unroll
    for (int i = 0; i < 8; ++i) { mrun[i] = -1e30f; lrun[i] = 0.f; }
    v8f o0 = {}, o1 = {}, o2 = {}, o3 = {};

    unsigned short* pw = &pbuf[wvid][0];

    auto issue = [&](int par, int kj) {
        // K^T tile: rows d=0..63, 32 contiguous kj each (row stride SEQ)
        tdm_load_tile_2d((unsigned)(uintptr_t)&kt_s[par][0], ktb + kj, 32, 64, SEQ);
        // V tile: rows kj..kj+31, 64 contiguous d each (row stride DH)
        tdm_load_tile_2d((unsigned)(uintptr_t)&v_s[par][0],
                         vbase + (size_t)kj * DH, 64, 32, DH);
    };

    if (threadIdx.x < 32) issue(0, 0);

    for (int it = 0; it < SEQ / 32; ++it) {
        const int kj0 = it * 32;
        if (threadIdx.x < 32) {
            if (it + 1 < SEQ / 32) {
                issue((it + 1) & 1, kj0 + 32);
                __builtin_amdgcn_s_wait_tensorcnt((short)2);
            } else {
                __builtin_amdgcn_s_wait_tensorcnt((short)0);
            }
        }
        __syncthreads();
        const int par = it & 1;

        // ---- scores: S = Q * K^T (two 16x16 column tiles) ----
        v8f c0 = {}, c1 = {};
        Frag b;
        load_b(b, &kt_s[par][(0  + halfSel * 8) * 32 + 0  + lrow], 32);  c0 = bfwmma(aq0, b, c0);
        load_b(b, &kt_s[par][(32 + halfSel * 8) * 32 + 0  + lrow], 32);  c0 = bfwmma(aq1, b, c0);
        load_b(b, &kt_s[par][(0  + halfSel * 8) * 32 + 16 + lrow], 32);  c1 = bfwmma(aq0, b, c1);
        load_b(b, &kt_s[par][(32 + halfSel * 8) * 32 + 16 + lrow], 32);  c1 = bfwmma(aq1, b, c1);

        const int col0 = kj0 + lrow;
        const bool ok0 = amask[bi * SEQ + col0] != 0;
        const bool ok1 = amask[bi * SEQ + col0 + 16] != 0;

        // ---- online softmax update (per C-fragment row group) ----
#pragma unroll
        for (int i = 0; i < 8; ++i) {
            float s0 = ok0 ? c0[i] * 0.125f : -1e30f;
            float s1 = ok1 ? c1[i] * 0.125f : -1e30f;
            float tm = fmaxf(s0, s1);
#pragma unroll
            for (int off = 8; off; off >>= 1) tm = fmaxf(tm, __shfl_xor(tm, off, 32));
            const float mnew = fmaxf(mrun[i], tm);
            const float scale = __expf(mrun[i] - mnew);
            const float e0 = __expf(s0 - mnew);
            const float e1 = __expf(s1 - mnew);
            float rs = e0 + e1;
#pragma unroll
            for (int off = 8; off; off >>= 1) rs += __shfl_xor(rs, off, 32);
            lrun[i] = lrun[i] * scale + rs;
            mrun[i] = mnew;
            o0[i] *= scale; o1[i] *= scale; o2[i] *= scale; o3[i] *= scale;
            pw[(i + halfSel * 8) * 32 + lrow]      = f2bf(e0);
            pw[(i + halfSel * 8) * 32 + 16 + lrow] = f2bf(e1);
        }

        // re-swizzle P from C-layout to A-layout via wave-private LDS
        asm volatile("s_wait_dscnt 0x0" ::: "memory");
        __builtin_amdgcn_wave_barrier();
        Frag ap;
        load_a(ap, pw + lrow * 32 + halfSel * 8);
        __builtin_amdgcn_wave_barrier();

        // ---- O += P * V ----
        Frag bvf;
        load_b(bvf, &v_s[par][halfSel * 8 * 64 + 0  + lrow], 64); o0 = bfwmma(ap, bvf, o0);
        load_b(bvf, &v_s[par][halfSel * 8 * 64 + 16 + lrow], 64); o1 = bfwmma(ap, bvf, o1);
        load_b(bvf, &v_s[par][halfSel * 8 * 64 + 32 + lrow], 64); o2 = bfwmma(ap, bvf, o2);
        load_b(bvf, &v_s[par][halfSel * 8 * 64 + 48 + lrow], 64); o3 = bfwmma(ap, bvf, o3);
        __syncthreads();
    }

#pragma unroll
    for (int i = 0; i < 8; ++i) {
        const int r = qi0 + i + halfSel * 8;              // s within sequence
        const float inv = 1.f / lrun[i];
        const size_t rowoff = ((size_t)bi * SEQ + r) * H + hh * DH;
        ao[rowoff + 0  + lrow] = f2bf(o0[i] * inv);
        ao[rowoff + 16 + lrow] = f2bf(o1[i] * inv);
        ao[rowoff + 32 + lrow] = f2bf(o2[i] * inv);
        ao[rowoff + 48 + lrow] = f2bf(o3[i] * inv);
    }
}

// ---------------- output projection with TDM staging ----------------

__global__ __launch_bounds__(256) void k_proj(
    const unsigned short* __restrict__ ah, const unsigned short* __restrict__ wo,
    const float* __restrict__ bo, float* __restrict__ out)
{
    __shared__ unsigned short wtile[2][32 * 16];      // 2 KB
    const int lane = threadIdx.x & 31;
    const int wvid = threadIdx.x >> 5;
    const int mblk = blockIdx.x >> 6;
    const int nt = blockIdx.x & 63;
    const int m0 = (mblk * 8 + wvid) * 16;
    const int n0 = nt * 16;
    const int lrow = lane & 15;
    const int halfSel = (lane >> 4) & 1;
    const int arow = m0 + lrow;

    v8f c = {};
    if (threadIdx.x < 32)
        tdm_load_tile_2d((unsigned)(uintptr_t)&wtile[0][0], wo + n0, 16, 32, H);

    for (int it = 0; it < 32; ++it) {
        const int k0 = it * 32;
        if (threadIdx.x < 32) {
            if (it + 1 < 32) {
                tdm_load_tile_2d((unsigned)(uintptr_t)&wtile[(it + 1) & 1][0],
                                 wo + (size_t)(k0 + 32) * H + n0, 16, 32, H);
                __builtin_amdgcn_s_wait_tensorcnt((short)1);
            } else {
                __builtin_amdgcn_s_wait_tensorcnt((short)0);
            }
        }
        __syncthreads();
        Frag a, b;
        load_a(a, ah + (size_t)arow * H + k0 + halfSel * 8);
        load_b(b, &wtile[it & 1][halfSel * 8 * 16 + lrow], 16);
        c = bfwmma(a, b, c);
        __syncthreads();
    }
    const int col = n0 + lrow;
    const float bias = bo[col];
#pragma unroll
    for (int i = 0; i < 8; ++i) {
        const int r = m0 + i + halfSel * 8;
        out[(size_t)r * H + col] = c[i] + bias;
    }
}

// ---------------- host launcher ----------------

extern "C" void kernel_launch(void* const* d_in, const int* in_sizes, int n_in,
                              void* d_out, int out_size, void* d_ws, size_t ws_size,
                              hipStream_t stream) {
    const float* x      = (const float*)d_in[0];
    const float* rmask  = (const float*)d_in[1];
    const int*   amask  = (const int*)d_in[2];
    const float* Wq  = (const float*)d_in[3];  const float* bq  = (const float*)d_in[4];
    const float* Wk  = (const float*)d_in[5];  const float* bk  = (const float*)d_in[6];
    const float* Wv  = (const float*)d_in[7];  const float* bv  = (const float*)d_in[8];
    const float* Wo  = (const float*)d_in[9];  const float* bo  = (const float*)d_in[10];
    const float* Wqr = (const float*)d_in[11]; const float* bqr = (const float*)d_in[12];
    const float* Wkr = (const float*)d_in[13]; const float* bkr = (const float*)d_in[14];
    float* out = (float*)d_out;

    char* ws = (char*)d_ws;
    size_t off = 0;
    auto carve = [&](size_t bytes) { char* p = ws + off; off += (bytes + 255) & ~(size_t)255; return p; };
    unsigned short* xh   = (unsigned short*)carve((size_t)BS * H * 2);
    unsigned short* wtq  = (unsigned short*)carve((size_t)H * H * 2);
    unsigned short* wtk  = (unsigned short*)carve((size_t)H * H * 2);
    unsigned short* wtv  = (unsigned short*)carve((size_t)H * H * 2);
    unsigned short* wtqr = (unsigned short*)carve((size_t)H * H * 2);
    unsigned short* wtkr = (unsigned short*)carve((size_t)H * H * 2);
    unsigned short* wto  = (unsigned short*)carve((size_t)H * H * 2);
    unsigned short* qbuf = (unsigned short*)carve((size_t)BATCH * HEADS * SEQ * DH * 2);
    unsigned short* ktb  = (unsigned short*)carve((size_t)BATCH * HEADS * SEQ * DH * 2);
    unsigned short* vbuf = (unsigned short*)carve((size_t)BATCH * HEADS * SEQ * DH * 2);
    unsigned short* aob  = (unsigned short*)carve((size_t)BS * H * 2);

    k_cvt_x<<<(BS * H + 255) / 256, 256, 0, stream>>>(x, xh, BS * H);
    k_cvt_w<<<H, 256, 0, stream>>>(Wq,  wtq);
    k_cvt_w<<<H, 256, 0, stream>>>(Wk,  wtk);
    k_cvt_w<<<H, 256, 0, stream>>>(Wv,  wtv);
    k_cvt_w<<<H, 256, 0, stream>>>(Wqr, wtqr);
    k_cvt_w<<<H, 256, 0, stream>>>(Wkr, wtkr);
    k_cvt_w<<<H, 256, 0, stream>>>(Wo,  wto);

    // (BS/128) * (H/16) = 32*64 = 2048 blocks, 8 waves each
    k_qkv<<<2048, 256, 0, stream>>>(xh, wtq, wtk, wtv, wtqr, wtkr,
                                    bq, bk, bv, bqr, bkr, rmask,
                                    qbuf, ktb, vbuf);

    // B*HEADS*(SEQ/128) = 512 blocks, 8 waves each
    k_attn<<<512, 256, 0, stream>>>(qbuf, ktb, vbuf, amask, aob);

    k_proj<<<2048, 256, 0, stream>>>(aob, wto, bo, out);
}
